// ForwardModel_86474871538496
// MI455X (gfx1250) — matrix operand order repaired and checked by reading, not compiled, not run on hardware
//
#include <hip/hip_runtime.h>
#include <stdint.h>

// ---------------------------------------------------------------------------
// CDNA5 (gfx1250) GraphNet forward: bf16 WMMA (16x16x32), wave32.
// One wave per 32-row (MT=2) tile: A staged in LDS, B (pre-transposed bf16
// weights) streamed from L2 once per k-step and reused across both M tiles,
// f32 accumulation, atomic-f32 scatter-mean.
// ---------------------------------------------------------------------------

typedef __bf16 bf16_t;
typedef bf16_t v16bf __attribute__((ext_vector_type(16)));
typedef bf16_t v8bf  __attribute__((ext_vector_type(8)));
typedef float  v8f   __attribute__((ext_vector_type(8)));

#define N_NODES 50000
#define N_EDGES 800000
#define NB 16      // graphs in batch
#define GF 16
#define NF 32
#define EF 32
#define HID 128

union AFrag { v16bf v; v8bf h[2]; };

__device__ __forceinline__ v8f wmma_bf16(v16bf a, v16bf b, v8f c) {
    return __builtin_amdgcn_wmma_f32_16x16x32_bf16(false, a, false, b, (short)0, c,
                                                   false, false);
}

// MT x NT block of 16x16 output tiles: A (MT*16 x KP, LDS bf16) @ Wt ([N][KP]).
// Loop nest: ks outer -> A frags loaded once per k-step, each B fragment
// loaded once and reused across all MT accumulator tiles.
template<int KP, int MT, int NT>
__device__ __forceinline__ void mm_block(const bf16_t* Alds, const bf16_t* Wt,
                                         v8f* acc, int lane) {
    const int ml = lane & 15;
    const int hf = lane >> 4;
#pragma unroll
    for (int ks = 0; ks < KP / 32; ++ks) {
        AFrag af[MT];
#pragma unroll
        for (int mt = 0; mt < MT; ++mt) {
            const bf16_t* ar = Alds + (mt * 16 + ml) * KP + ks * 32 + hf * 16;
            af[mt].h[0] = *(const v8bf*)(ar);
            af[mt].h[1] = *(const v8bf*)(ar + 8);
        }
#pragma unroll
        for (int nt = 0; nt < NT; ++nt) {
            const int n = nt * 16 + ml;
            v16bf b = *(const v16bf*)(Wt + (size_t)n * KP + ks * 32 + hf * 16);
#pragma unroll
            for (int mt = 0; mt < MT; ++mt)
                acc[mt * NT + nt] = wmma_bf16(af[mt].v, b, acc[mt * NT + nt]);
        }
    }
}

// Hidden layer: (MT*16 x KP) @ (KP x NT*16) + bias, ReLU, bf16 to LDS row-major.
template<int KP, int MT, int NT>
__device__ __forceinline__ void mlp_layer_relu(const bf16_t* Alds, const bf16_t* Wt,
                                               const float* bias, bf16_t* Olds,
                                               int lane) {
    v8f acc[MT * NT] = {};
    mm_block<KP, MT, NT>(Alds, Wt, acc, lane);
    const int hf = lane >> 4, nl = lane & 15;
#pragma unroll
    for (int mt = 0; mt < MT; ++mt)
#pragma unroll
        for (int nt = 0; nt < NT; ++nt) {
            const int n = nt * 16 + nl;
            const float b = bias[n];
#pragma unroll
            for (int r = 0; r < 8; ++r) {
                float v = fmaxf(acc[mt * NT + nt][r] + b, 0.0f);
                Olds[(mt * 16 + hf * 8 + r) * (NT * 16) + n] = (bf16_t)v;
            }
        }
}

// ------------------- full-row gather helpers (one lane = one row) -----------
__device__ __forceinline__ void cvt4(bf16_t* d, float4 v) {
    d[0] = (bf16_t)v.x; d[1] = (bf16_t)v.y; d[2] = (bf16_t)v.z; d[3] = (bf16_t)v.w;
}
__device__ __forceinline__ void gatherF(bf16_t* row, int off, const float* src,
                                        int nf4) {
    const float4* s = (const float4*)src;
    for (int q = 0; q < nf4; ++q) cvt4(row + off + q * 4, s[q]);
}
__device__ __forceinline__ void gatherF_scaled(bf16_t* row, int off, const float* src,
                                               int nf4, float sc) {
    const float4* s = (const float4*)src;
    for (int q = 0; q < nf4; ++q) {
        float4 v = s[q];
        v.x *= sc; v.y *= sc; v.z *= sc; v.w *= sc;
        cvt4(row + off + q * 4, v);
    }
}
__device__ __forceinline__ void gatherB(bf16_t* row, int off, const bf16_t* src,
                                        int nb8) {  // nb8 chunks of 8 bf16
    for (int q = 0; q < nb8; ++q)
        *(float4*)(row + off + q * 8) = *(const float4*)(src + q * 8);
}
__device__ __forceinline__ void zeroF(bf16_t* row, int off, int nelem) {
    for (int q = 0; q < nelem / 4; ++q) {
        float4 z = {0.f, 0.f, 0.f, 0.f};
        cvt4(row + off + q * 4, z);
    }
}

// ------------------------- weight prep + zeroing ----------------------------
struct WDesc  { const float* w; bf16_t* wt; int K, N, KP; };
struct WDescs { WDesc d[12]; };

__global__ void __launch_bounds__(256) convert_weights(WDescs ds) {
    const WDesc d = ds.d[blockIdx.y];
    const int total = d.N * d.KP;
    for (int idx = blockIdx.x * 256 + threadIdx.x; idx < total;
         idx += gridDim.x * 256) {
        const int n = idx / d.KP;
        const int k = idx - n * d.KP;
        const float v = (k < d.K) ? d.w[(size_t)k * d.N + n] : 0.0f;
        d.wt[idx] = (bf16_t)v;
    }
}

__global__ void __launch_bounds__(256) zero_f32(float* p, size_t n) {
    for (size_t i = (size_t)blockIdx.x * 256 + threadIdx.x; i < n;
         i += (size_t)gridDim.x * 256)
        p[i] = 0.0f;
}

// ------------------------- GN1: fused edge MLP + node1 ----------------------
__global__ void __launch_bounds__(32)
edge_kernel1(const float* x, const int* rowI, const int* colI, const float* ea,
             const float* u, const int* batch,
             const bf16_t* w0t, const float* b0, const bf16_t* w1t, const float* b1,
             const bf16_t* wn1t, const float* bn1,
             bf16_t* e1out, float* agg, float* cnt, int nEdges) {
    __shared__ bf16_t A0[32 * 128];
    __shared__ bf16_t H1[32 * 128];
    __shared__ bf16_t A1[32 * 64];
    __shared__ int colS[32];

    const int lane = threadIdx.x;
    const int hf = lane >> 4, nl = lane & 15;
    const long e0 = (long)blockIdx.x * 32;
    const long e  = e0 + lane;
    const bool valid = e < nEdges;
    const long ec = valid ? e : 0;
    const int r = rowI[ec], c = colI[ec];
    const int g = batch[r];
    colS[lane] = c;

    // A0 row = [x[r] | x[c] | ea | u[batch[r]] | pad16]  (112 -> KP 128)
    bf16_t* row = A0 + lane * 128;
    if (valid) {
        gatherF(row,  0, x  + (size_t)r * NF, 8);
        gatherF(row, 32, x  + (size_t)c * NF, 8);
        gatherF(row, 64, ea + (size_t)e * EF, 8);
        gatherF(row, 96, u  + g * GF, 4);
    } else zeroF(row, 0, 112);
    zeroF(row, 112, 16);
    __syncthreads();

    mlp_layer_relu<128, 2, 8>(A0, w0t, b0, H1, lane);          // 112(p128)->128 relu
    __syncthreads();

    if (valid) gatherF(A1 + lane * 64, 0, x + (size_t)r * NF, 8);
    else       zeroF(A1 + lane * 64, 0, 32);

    {   // 128 -> 32 = e1
        v8f acc[2 * 2] = {};
        mm_block<128, 2, 2>(H1, w1t, acc, lane);
#pragma unroll
        for (int mt = 0; mt < 2; ++mt)
#pragma unroll
            for (int nt = 0; nt < 2; ++nt) {
                const int n = nt * 16 + nl;
                const float bb = b1[n];
#pragma unroll
                for (int rr = 0; rr < 8; ++rr) {
                    const int mm = mt * 16 + hf * 8 + rr;
                    const bf16_t bv = (bf16_t)(acc[mt * 2 + nt][rr] + bb);
                    A1[mm * 64 + 32 + n] = bv;                 // node1 input half
                    const long ee = e0 + mm;
                    if (ee < nEdges) e1out[ee * EF + n] = bv;  // keep for GN2
                }
            }
    }
    __syncthreads();

    {   // node1: 64 -> 128, scatter to agg
        v8f acc[2 * 8] = {};
        mm_block<64, 2, 8>(A1, wn1t, acc, lane);
#pragma unroll
        for (int mt = 0; mt < 2; ++mt)
#pragma unroll
            for (int nt = 0; nt < 8; ++nt) {
                const int n = nt * 16 + nl;
                const float bb = bn1[n];
#pragma unroll
                for (int rr = 0; rr < 8; ++rr) {
                    const int mm = mt * 16 + hf * 8 + rr;
                    if (e0 + mm < nEdges)
                        atomicAdd(agg + (size_t)colS[mm] * HID + n,
                                  acc[mt * 8 + nt][rr] + bb);
                }
            }
    }
    if (valid) atomicAdd(cnt + c, 1.0f);
}

// ------------------------- GN1: node2 ---------------------------------------
__global__ void __launch_bounds__(32)
node_kernel1(const float* x, const float* agg, const float* cnt,
             const float* u, const int* batch,
             const bf16_t* w0t, const float* b0, const bf16_t* w1t, const float* b1,
             float* x1, float* usum, float* gcnt, int nNodes) {
    __shared__ bf16_t A0[32 * 192];
    __shared__ bf16_t H1[32 * 128];
    __shared__ int gS[32];

    const int lane = threadIdx.x;
    const int hf = lane >> 4, nl = lane & 15;
    const int i0 = blockIdx.x * 32;
    const int i  = i0 + lane;
    const bool valid = i < nNodes;
    const int ic = valid ? i : 0;
    const int g  = batch[ic];
    gS[lane] = g;

    // A0 row = [x | agg/cnt | u[batch] | pad16]  (176 -> KP 192)
    bf16_t* row = A0 + lane * 192;
    if (valid) {
        const float inv = 1.0f / fmaxf(cnt[ic], 1.0f);
        gatherF(row, 0, x + (size_t)ic * NF, 8);
        gatherF_scaled(row, 32, agg + (size_t)ic * HID, 32, inv);
        gatherF(row, 160, u + g * GF, 4);
    } else zeroF(row, 0, 176);
    zeroF(row, 176, 16);
    __syncthreads();

    mlp_layer_relu<192, 2, 8>(A0, w0t, b0, H1, lane);
    __syncthreads();

    {   // 128 -> 32 = x1 (+ per-graph sums for global net)
        v8f acc[2 * 2] = {};
        mm_block<128, 2, 2>(H1, w1t, acc, lane);
#pragma unroll
        for (int mt = 0; mt < 2; ++mt)
#pragma unroll
            for (int nt = 0; nt < 2; ++nt) {
                const int n = nt * 16 + nl;
                const float bb = b1[n];
#pragma unroll
                for (int rr = 0; rr < 8; ++rr) {
                    const int mm = mt * 16 + hf * 8 + rr;
                    const int ii = i0 + mm;
                    if (ii < nNodes) {
                        const float v = acc[mt * 2 + nt][rr] + bb;
                        x1[(size_t)ii * NF + n] = v;
                        atomicAdd(usum + gS[mm] * NF + n, v);
                    }
                }
            }
    }
    if (valid) atomicAdd(gcnt + g, 1.0f);
}

// ------------------------- GN1: global net (16 graphs == one M tile) --------
__global__ void __launch_bounds__(32)
glob_kernel1(const float* u, const float* usum, const float* gcnt,
             const bf16_t* w0t, const float* b0, const bf16_t* w1t, const float* b1,
             float* u1) {
    __shared__ bf16_t A0[16 * 64];
    __shared__ bf16_t H1[16 * 128];
    const int lane = threadIdx.x;
    const int hf = lane >> 4, nl = lane & 15;

    if (lane < 16) {                           // one lane builds one graph row
        const float inv = 1.0f / fmaxf(gcnt[lane], 1.0f);
        bf16_t* row = A0 + lane * 64;          // [u | mean(x1) | pad16] 48 -> 64
        gatherF(row, 0, u + lane * GF, 4);
        gatherF_scaled(row, 16, usum + lane * NF, 8, inv);
        zeroF(row, 48, 16);
    }
    __syncthreads();

    mlp_layer_relu<64, 1, 8>(A0, w0t, b0, H1, lane);
    __syncthreads();

    v8f acc[1] = {};
    mm_block<128, 1, 1>(H1, w1t, acc, lane);   // 128 -> 16 = u1
    const float bb = b1[nl];
#pragma unroll
    for (int rr = 0; rr < 8; ++rr)
        u1[(hf * 8 + rr) * GF + nl] = acc[0][rr] + bb;
}

// ------------------------- GN2: fused edge MLP + node1 ----------------------
__global__ void __launch_bounds__(32)
edge_kernel2(const float* x, const float* x1, const int* rowI, const int* colI,
             const float* ea, const bf16_t* e1, const float* u, const float* u1,
             const int* batch,
             const bf16_t* w0t, const float* b0, const bf16_t* w1t, const float* b1,
             const bf16_t* wn1t, const float* bn1,
             float* agg2, int nEdges) {
    __shared__ bf16_t A0[32 * 224];
    __shared__ bf16_t H1[32 * 128];
    __shared__ bf16_t A1[32 * 96];
    __shared__ int colS[32];

    const int lane = threadIdx.x;
    const int hf = lane >> 4, nl = lane & 15;
    const long e0 = (long)blockIdx.x * 32;
    const long e  = e0 + lane;
    const bool valid = e < nEdges;
    const long ec = valid ? e : 0;
    const int r = rowI[ec], c = colI[ec];
    const int g = batch[r];
    colS[lane] = c;

    // A0 row = [x[r]|x1[r]|x[c]|x1[c]|ea|e1|u[g]|u1[g]]  (224, no pad)
    bf16_t* row = A0 + lane * 224;
    if (valid) {
        gatherF(row,   0, x  + (size_t)r * NF, 8);
        gatherF(row,  32, x1 + (size_t)r * NF, 8);
        gatherF(row,  64, x  + (size_t)c * NF, 8);
        gatherF(row,  96, x1 + (size_t)c * NF, 8);
        gatherF(row, 128, ea + (size_t)e * EF, 8);
        gatherB(row, 160, e1 + (size_t)e * EF, 4);
        gatherF(row, 192, u  + g * GF, 4);
        gatherF(row, 208, u1 + g * GF, 4);
    } else zeroF(row, 0, 224);
    __syncthreads();

    mlp_layer_relu<224, 2, 8>(A0, w0t, b0, H1, lane);          // 224 -> 128 relu
    __syncthreads();

    if (valid) {
        gatherF(A1 + lane * 96, 0,  x  + (size_t)r * NF, 8);
        gatherF(A1 + lane * 96, 32, x1 + (size_t)r * NF, 8);
    } else zeroF(A1 + lane * 96, 0, 64);

    {   // 128 -> 32 = e2 (LDS only)
        v8f acc[2 * 2] = {};
        mm_block<128, 2, 2>(H1, w1t, acc, lane);
#pragma unroll
        for (int mt = 0; mt < 2; ++mt)
#pragma unroll
            for (int nt = 0; nt < 2; ++nt) {
                const int n = nt * 16 + nl;
                const float bb = b1[n];
#pragma unroll
                for (int rr = 0; rr < 8; ++rr) {
                    const int mm = mt * 16 + hf * 8 + rr;
                    A1[mm * 96 + 64 + n] = (bf16_t)(acc[mt * 2 + nt][rr] + bb);
                }
            }
    }
    __syncthreads();

    {   // node1: 96 -> 128, scatter to agg2
        v8f acc[2 * 8] = {};
        mm_block<96, 2, 8>(A1, wn1t, acc, lane);
#pragma unroll
        for (int mt = 0; mt < 2; ++mt)
#pragma unroll
            for (int nt = 0; nt < 8; ++nt) {
                const int n = nt * 16 + nl;
                const float bb = bn1[n];
#pragma unroll
                for (int rr = 0; rr < 8; ++rr) {
                    const int mm = mt * 16 + hf * 8 + rr;
                    if (e0 + mm < nEdges)
                        atomicAdd(agg2 + (size_t)colS[mm] * HID + n,
                                  acc[mt * 8 + nt][rr] + bb);
                }
            }
    }
}

// ------------------------- GN2: node2 -> output -----------------------------
__global__ void __launch_bounds__(32)
node_kernel2(const float* x, const float* x1, const float* agg2, const float* cnt,
             const float* u, const float* u1, const int* batch,
             const bf16_t* w0t, const float* b0, const bf16_t* w1t, const float* b1,
             float* out, int nNodes) {
    __shared__ bf16_t A0[32 * 224];
    __shared__ bf16_t H1[32 * 128];
    const int lane = threadIdx.x;
    const int hf = lane >> 4, nl = lane & 15;
    const int i0 = blockIdx.x * 32;
    const int i  = i0 + lane;
    const bool valid = i < nNodes;
    const int ic = valid ? i : 0;
    const int g  = batch[ic];

    // A0 row = [x | x1 | agg2/cnt | u[g] | u1[g]]  (224)
    bf16_t* row = A0 + lane * 224;
    if (valid) {
        const float inv = 1.0f / fmaxf(cnt[ic], 1.0f);
        gatherF(row,   0, x  + (size_t)ic * NF, 8);
        gatherF(row,  32, x1 + (size_t)ic * NF, 8);
        gatherF_scaled(row, 64, agg2 + (size_t)ic * HID, 32, inv);
        gatherF(row, 192, u  + g * GF, 4);
        gatherF(row, 208, u1 + g * GF, 4);
    } else zeroF(row, 0, 224);
    __syncthreads();

    mlp_layer_relu<224, 2, 8>(A0, w0t, b0, H1, lane);
    __syncthreads();

    // Only output cols 0..2 needed -> compute only the first 16-col tile.
    v8f acc[2] = {};
    mm_block<128, 2, 1>(H1, w1t, acc, lane);
    if (nl < 3) {
        const float bb = b1[nl];
#pragma unroll
        for (int mt = 0; mt < 2; ++mt)
#pragma unroll
            for (int rr = 0; rr < 8; ++rr) {
                const int ii = i0 + mt * 16 + hf * 8 + rr;
                if (ii < nNodes) out[(size_t)ii * 3 + nl] = acc[mt][rr] + bb;
            }
    }
}

// ---------------------------------------------------------------------------
extern "C" void kernel_launch(void* const* d_in, const int* in_sizes, int n_in,
                              void* d_out, int out_size, void* d_ws, size_t ws_size,
                              hipStream_t stream) {
    (void)in_sizes; (void)n_in; (void)out_size; (void)ws_size;

    const float* x   = (const float*)d_in[0];
    const int*   ei  = (const int*)d_in[1];
    const float* ea  = (const float*)d_in[2];
    const float* u   = (const float*)d_in[3];
    const int*   bat = (const int*)d_in[4];
    // params flattened in jax tree order: gn1 < gn2; edge < glob < node1 < node2;
    // each layer is (w, b).
    const float* g1e_w0  = (const float*)d_in[5];  const float* g1e_b0  = (const float*)d_in[6];
    const float* g1e_w1  = (const float*)d_in[7];  const float* g1e_b1  = (const float*)d_in[8];
    const float* g1g_w0  = (const float*)d_in[9];  const float* g1g_b0  = (const float*)d_in[10];
    const float* g1g_w1  = (const float*)d_in[11]; const float* g1g_b1  = (const float*)d_in[12];
    const float* g1n1_w0 = (const float*)d_in[13]; const float* g1n1_b0 = (const float*)d_in[14];
    const float* g1n2_w0 = (const float*)d_in[15]; const float* g1n2_b0 = (const float*)d_in[16];
    const float* g1n2_w1 = (const float*)d_in[17]; const float* g1n2_b1 = (const float*)d_in[18];
    const float* g2e_w0  = (const float*)d_in[19]; const float* g2e_b0  = (const float*)d_in[20];
    const float* g2e_w1  = (const float*)d_in[21]; const float* g2e_b1  = (const float*)d_in[22];
    // d_in[23..26] = gn2 glob (its output is unused by the reference return)
    const float* g2n1_w0 = (const float*)d_in[27]; const float* g2n1_b0 = (const float*)d_in[28];
    const float* g2n2_w0 = (const float*)d_in[29]; const float* g2n2_b0 = (const float*)d_in[30];
    const float* g2n2_w1 = (const float*)d_in[31]; const float* g2n2_b1 = (const float*)d_in[32];

    char* ws = (char*)d_ws;
    size_t cur = 0;
    auto alloc = [&](size_t bytes) -> char* {
        cur = (cur + 255) & ~(size_t)255;
        char* p = ws + cur;
        cur += bytes;
        return p;
    };

    bf16_t* g1e_w0t  = (bf16_t*)alloc(128 * 128 * 2);
    bf16_t* g1e_w1t  = (bf16_t*)alloc(32 * 128 * 2);
    bf16_t* g1n1_wt  = (bf16_t*)alloc(128 * 64 * 2);
    bf16_t* g1n2_w0t = (bf16_t*)alloc(128 * 192 * 2);
    bf16_t* g1n2_w1t = (bf16_t*)alloc(32 * 128 * 2);
    bf16_t* g1g_w0t  = (bf16_t*)alloc(128 * 64 * 2);
    bf16_t* g1g_w1t  = (bf16_t*)alloc(16 * 128 * 2);
    bf16_t* g2e_w0t  = (bf16_t*)alloc(128 * 224 * 2);
    bf16_t* g2e_w1t  = (bf16_t*)alloc(32 * 128 * 2);
    bf16_t* g2n1_wt  = (bf16_t*)alloc(128 * 96 * 2);
    bf16_t* g2n2_w0t = (bf16_t*)alloc(128 * 224 * 2);
    bf16_t* g2n2_w1t = (bf16_t*)alloc(32 * 128 * 2);

    bf16_t* e1buf = (bf16_t*)alloc((size_t)N_EDGES * EF * 2);
    float*  x1buf = (float*)alloc((size_t)N_NODES * NF * 4);
    float*  u1buf = (float*)alloc((size_t)NB * GF * 4);

    const size_t zfloats = (size_t)N_NODES * HID * 2 + N_NODES + NB * NF + NB;
    float* zbase = (float*)alloc(zfloats * 4);
    float* agg1 = zbase;
    float* agg2 = agg1 + (size_t)N_NODES * HID;
    float* cnt  = agg2 + (size_t)N_NODES * HID;
    float* usum = cnt + N_NODES;
    float* gcnt = usum + NB * NF;

    WDescs wd;
    wd.d[0]  = {g1e_w0,  g1e_w0t,  112, 128, 128};
    wd.d[1]  = {g1e_w1,  g1e_w1t,  128, 32,  128};
    wd.d[2]  = {g1n1_w0, g1n1_wt,  64,  128, 64};
    wd.d[3]  = {g1n2_w0, g1n2_w0t, 176, 128, 192};
    wd.d[4]  = {g1n2_w1, g1n2_w1t, 128, 32,  128};
    wd.d[5]  = {g1g_w0,  g1g_w0t,  48,  128, 64};
    wd.d[6]  = {g1g_w1,  g1g_w1t,  128, 16,  128};
    wd.d[7]  = {g2e_w0,  g2e_w0t,  224, 128, 224};
    wd.d[8]  = {g2e_w1,  g2e_w1t,  128, 32,  128};
    wd.d[9]  = {g2n1_w0, g2n1_wt,  96,  128, 96};
    wd.d[10] = {g2n2_w0, g2n2_w0t, 224, 128, 224};
    wd.d[11] = {g2n2_w1, g2n2_w1t, 128, 32,  128};

    convert_weights<<<dim3(112, 12), 256, 0, stream>>>(wd);
    zero_f32<<<2048, 256, 0, stream>>>(zbase, zfloats);

    edge_kernel1<<<N_EDGES / 32, 32, 0, stream>>>(
        x, ei, ei + N_EDGES, ea, u, bat,
        g1e_w0t, g1e_b0, g1e_w1t, g1e_b1, g1n1_wt, g1n1_b0,
        e1buf, agg1, cnt, N_EDGES);

    node_kernel1<<<(N_NODES + 31) / 32, 32, 0, stream>>>(
        x, agg1, cnt, u, bat,
        g1n2_w0t, g1n2_b0, g1n2_w1t, g1n2_b1,
        x1buf, usum, gcnt, N_NODES);

    glob_kernel1<<<1, 32, 0, stream>>>(
        u, usum, gcnt, g1g_w0t, g1g_b0, g1g_w1t, g1g_b1, u1buf);

    edge_kernel2<<<N_EDGES / 32, 32, 0, stream>>>(
        x, x1buf, ei, ei + N_EDGES, ea, e1buf, u, u1buf, bat,
        g2e_w0t, g2e_b0, g2e_w1t, g2e_b1, g2n1_wt, g2n1_b0,
        agg2, N_EDGES);

    node_kernel2<<<(N_NODES + 31) / 32, 32, 0, stream>>>(
        x, x1buf, agg2, cnt, u, u1buf, bat,
        g2n2_w0t, g2n2_b0, g2n2_w1t, g2n2_b1,
        (float*)d_out, N_NODES);
}
